// _CausalSelfAttention_56616258896393
// MI455X (gfx1250) — matrix operand order repaired and checked by reading, not compiled, not run on hardware
//
#include <hip/hip_runtime.h>

// ---------------------------------------------------------------------------
// CDNA5 (gfx1250) causal self-attention, bf16 WMMA + TDM-staged GEMM weights.
//   v_wmma_f32_16x16x32_bf16:  D(16x16 f32) = A(16x32 bf16) x B(32x16 bf16) + C
// Fragment layouts (ISA 7.12.2, wave32):
//   A: lane holds row M=lane&15;  lane<16: K{0..15},       lane>=16: K{16..31} (contig)
//   B: lane holds col N=lane&15;  lane<16: K{0..15},       lane>=16: K{16..31} (contig)
//   C: VGPR r, lanes 0-15 -> (M=r,   N=lane); lanes 16-31 -> (M=r+8, N=lane-16)
// GEMM B-tiles staged to LDS by the Tensor Data Mover (ISA ch.8), padded rows.
// ---------------------------------------------------------------------------

typedef __bf16 bf16_t;
typedef __bf16 v16bf __attribute__((ext_vector_type(16)));
typedef __bf16 v8bf  __attribute__((ext_vector_type(8)));
typedef __bf16 v4bf  __attribute__((ext_vector_type(4)));
typedef float  v8f   __attribute__((ext_vector_type(8)));
typedef unsigned int v4u __attribute__((ext_vector_type(4)));
typedef int    v8i   __attribute__((ext_vector_type(8)));
typedef int    v4i   __attribute__((ext_vector_type(4)));

#define WMMA_BF16(a, b, c) \
  __builtin_amdgcn_wmma_f32_16x16x32_bf16(false, (a), false, (b), (short)0, (c), false, false)

constexpr int B_ = 4, S_ = 1024, E_ = 1024, H_ = 16, HD_ = 64;
constexpr float LOG2E_ = 1.4426950408889634f;
constexpr float NEGF_  = -1e30f;

// LDS tile geometry: 128 rows x 32 bf16, TDM pad 4 DW after every 16 DW
// -> row stride 40 halfs (80B). ds_load_b128 lane banks (l*20 mod 64): no conflicts.
constexpr int TROW_ = 40;                    // halfs per padded LDS row
constexpr int TBUF_HALFS_ = 128 * TROW_;     // 10240B per buffer

__device__ __forceinline__ v8f zero_v8f() {
  v8f z;
#pragma unroll
  for (int i = 0; i < 8; ++i) z[i] = 0.f;
  return z;
}

// A fragment (16x32 bf16) from row-major [*, ld] at (row0=0, k0=0) == base
__device__ __forceinline__ v16bf load_a_frag(const bf16_t* base, int ld, int lane) {
  const int row = lane & 15, hi = (lane >> 4) & 1;
  const bf16_t* p = base + (long long)row * ld + hi * 16;
  v8bf c0 = *(const v8bf*)(p);        // K = hi*16 .. hi*16+7
  v8bf c1 = *(const v8bf*)(p + 8);    // K = hi*16+8 .. hi*16+15
  v16bf r;
#pragma unroll
  for (int i = 0; i < 8; ++i) { r[i] = c0[i]; r[i + 8] = c1[i]; }
  return r;
}

// -------- Tensor Data Mover: 2D tile (32 x 128 rows) global -> LDS ---------
// D# fields per ISA 8.3/8.4. data_size=2B, pad_enable, pad_interval=16DW(code 3),
// pad_amount=4DW(code 3). tile_dim0=32 elems (64B rows), tile_dim1=128 rows.
__device__ __forceinline__ void tdm_load_tile(unsigned int lds_byte_addr,
                                              const bf16_t* gptr,
                                              unsigned int tensor_rows) {
  const unsigned long long ga = (unsigned long long)(size_t)gptr;
  v4u g0;
  g0[0] = 1u;                                          // count=1, user mode
  g0[1] = lds_byte_addr;                               // lds_addr (bytes)
  g0[2] = (unsigned int)ga;                            // global_addr[31:0]
  g0[3] = (unsigned int)((ga >> 32) & 0x1FFFFFFull) |  // global_addr[56:32]
          (2u << 30);                                  // type=2 ("image")
  v8i g1;
  g1[0] = (int)((1u << 16) |        // data_size = 2 bytes
                (1u << 20) |        // pad_enable
                (3u << 22) |        // pad_interval: 16 DWORDs
                (3u << 25));        // pad_amount:   4 DWORDs
  g1[1] = (int)((1024u & 0xFFFFu) << 16);                      // tensor_dim0 lo16
  g1[2] = (int)(((1024u >> 16) & 0xFFFFu) |
                ((tensor_rows & 0xFFFFu) << 16));              // dim0 hi | dim1 lo
  g1[3] = (int)(((tensor_rows >> 16) & 0xFFFFu) | (32u << 16)); // dim1 hi | tile_dim0
  g1[4] = (int)(128u);                                         // tile_dim1 | tile_dim2=0
  g1[5] = (int)(1024);                                         // tensor_dim0_stride lo
  g1[6] = 0;
  g1[7] = 0;
  v4i z4 = {0, 0, 0, 0};
#if defined(__clang_major__) && __clang_major__ >= 23
  v8i z8 = {0, 0, 0, 0, 0, 0, 0, 0};
  __builtin_amdgcn_tensor_load_to_lds(g0, g1, z4, z4, z8, 0);
#else
  __builtin_amdgcn_tensor_load_to_lds(g0, g1, z4, z4, 0);
#endif
}

// ------------------------------- prologue casts ----------------------------

__global__ void k_cast_x(const float4* __restrict__ in, v4bf* __restrict__ out, int n4) {
  int i = blockIdx.x * blockDim.x + threadIdx.x;
  if (i < n4) {
    float4 v = in[i];
    v4bf o;
    o[0] = (bf16_t)v.x; o[1] = (bf16_t)v.y; o[2] = (bf16_t)v.z; o[3] = (bf16_t)v.w;
    out[i] = o;
  }
}

// in: [K, N] row-major f32  ->  out: [N, K] row-major bf16
__global__ void k_transpose_cast(const float* __restrict__ in, bf16_t* __restrict__ out,
                                 int K, int N) {
  const long long total = (long long)K * N;
  for (long long i = (long long)blockIdx.x * blockDim.x + threadIdx.x; i < total;
       i += (long long)gridDim.x * blockDim.x) {
    int n = (int)(i / K), k = (int)(i % K);
    out[i] = (bf16_t)in[(long long)k * N + n];
  }
}

// ------------------------------- QKV GEMM ----------------------------------
// C[4096,3072] = Xb[4096,1024] @ Wqkv + b ; routed into Q/K (bhsd) and V^T (bhds)
// 256 threads = 8 waves; block tile 128(M) x 128(N); B tile TDM-staged in LDS.
__global__ __launch_bounds__(256) void k_qkv_gemm(
    const bf16_t* __restrict__ Xb, const bf16_t* __restrict__ WT,
    const float* __restrict__ bias, bf16_t* __restrict__ Qp,
    bf16_t* __restrict__ Kp, bf16_t* __restrict__ Vt) {
  const int lane = threadIdx.x & 31, wave = threadIdx.x >> 5;
  const int ln = lane & 15, hi = lane >> 4;
  const int Mbase = blockIdx.y * 128 + wave * 16;
  const int Nbase = blockIdx.x * 128;

  __shared__ bf16_t Bsh[2][TBUF_HALFS_];
  const unsigned int lds_b[2] = {(unsigned int)(size_t)&Bsh[0][0],
                                 (unsigned int)(size_t)&Bsh[1][0]};
  const bf16_t* wtile = WT + (long long)Nbase * E_;

  v8f acc[8];
#pragma unroll
  for (int f = 0; f < 8; ++f) acc[f] = zero_v8f();

  if (wave == 0) tdm_load_tile(lds_b[0], wtile, 3u * E_);

  constexpr int NK = E_ / 32;
  for (int kt = 0; kt < NK; ++kt) {
    if (wave == 0) __builtin_amdgcn_s_wait_tensorcnt(0);
    __syncthreads();  // tile kt visible to all; prev buffer fully consumed
    if (wave == 0 && kt + 1 < NK)
      tdm_load_tile(lds_b[(kt + 1) & 1], wtile + (kt + 1) * 32, 3u * E_);

    v16bf a = load_a_frag(Xb + (long long)Mbase * E_ + kt * 32, E_, lane);
    const bf16_t* bsh = &Bsh[kt & 1][0];
#pragma unroll
    for (int f = 0; f < 8; ++f) {
      v16bf bfrag = *(const v16bf*)(bsh + (f * 16 + ln) * TROW_ + hi * 16);
      acc[f] = WMMA_BF16(a, bfrag, acc[f]);
    }
  }

  const float qscale = 0.125f * LOG2E_;  // 1/sqrt(64), exp2-domain fold
#pragma unroll
  for (int f = 0; f < 8; ++f) {
    const int n = Nbase + f * 16 + ln;
    const float bs = bias[n];
#pragma unroll
    for (int r = 0; r < 8; ++r) {
      const int row = Mbase + r + hi * 8;          // global token index
      const int bb = row >> 10, s = row & (S_ - 1);
      const float val = acc[f][r] + bs;
      if (n < E_) {
        const int h = n >> 6, hd = n & 63;
        Qp[(((long long)bb * H_ + h) * S_ + s) * HD_ + hd] = (bf16_t)(val * qscale);
      } else if (n < 2 * E_) {
        const int nn = n - E_, h = nn >> 6, hd = nn & 63;
        Kp[(((long long)bb * H_ + h) * S_ + s) * HD_ + hd] = (bf16_t)val;
      } else {
        const int nn = n - 2 * E_, h = nn >> 6, hd = nn & 63;
        Vt[(((long long)bb * H_ + h) * HD_ + hd) * S_ + s] = (bf16_t)val;  // transposed
      }
    }
  }
}

// ------------------------------- attention ---------------------------------
// grid = (S/64, B*H), 128 threads (4 waves); wave owns a 16-query tile.
__global__ __launch_bounds__(128) void k_attn(
    const bf16_t* __restrict__ Qp, const bf16_t* __restrict__ Kp,
    const bf16_t* __restrict__ Vt, const int* __restrict__ amask,
    bf16_t* __restrict__ AO) {
  const int lane = threadIdx.x & 31, wave = threadIdx.x >> 5;
  const int ln = lane & 15, hi = lane >> 4;
  const int bh = blockIdx.y, bb = bh >> 4, h = bh & 15;
  const int qbase = blockIdx.x * 64 + wave * 16;

  const bf16_t* Qh = Qp + (long long)bh * S_ * HD_;
  const bf16_t* Kh = Kp + (long long)bh * S_ * HD_;
  const bf16_t* Vh = Vt + (long long)bh * HD_ * S_;
  const int* mrow = amask + bb * S_;

  __shared__ bf16_t pshm[4][16][40];  // per-wave 16x32 P tile, padded rows

  const v16bf aQ0 = load_a_frag(Qh + (long long)qbase * HD_, HD_, lane);
  const v16bf aQ1 = load_a_frag(Qh + (long long)qbase * HD_ + 32, HD_, lane);

  v8f o0 = zero_v8f(), o1 = zero_v8f(), o2 = zero_v8f(), o3 = zero_v8f();
  float mr[8], lr[8];
#pragma unroll
  for (int r = 0; r < 8; ++r) { mr[r] = NEGF_; lr[r] = 0.f; }

  const int nblocks = (qbase + 16 + 31) >> 5;  // causal: keys 0 .. qbase+15
  for (int jb = 0; jb < nblocks; ++jb) {
    const int j0 = jb * 32;
    // ---- S = Q K^T (exp2 domain; scale folded into Q) --------------------
    v8f s0 = zero_v8f(), s1 = zero_v8f();
    {
      v16bf bk00 = *(const v16bf*)(Kh + (long long)(j0 + ln) * HD_ + hi * 16);
      v16bf bk10 = *(const v16bf*)(Kh + (long long)(j0 + 16 + ln) * HD_ + hi * 16);
      s0 = WMMA_BF16(aQ0, bk00, s0);
      s1 = WMMA_BF16(aQ0, bk10, s1);
      v16bf bk01 = *(const v16bf*)(Kh + (long long)(j0 + ln) * HD_ + 32 + hi * 16);
      v16bf bk11 = *(const v16bf*)(Kh + (long long)(j0 + 16 + ln) * HD_ + 32 + hi * 16);
      s0 = WMMA_BF16(aQ1, bk01, s0);
      s1 = WMMA_BF16(aQ1, bk11, s1);
    }
    // ---- mask + online softmax (rows live across 16-lane halves) ---------
    const int key0 = j0 + ln, key1 = j0 + 16 + ln;
    const bool km0 = mrow[key0] != 0, km1 = mrow[key1] != 0;
#pragma unroll
    for (int r = 0; r < 8; ++r) {
      const int row = qbase + r + hi * 8;
      const float x0 = (km0 && key0 <= row) ? s0[r] : NEGF_;
      const float x1 = (km1 && key1 <= row) ? s1[r] : NEGF_;
      float v = fmaxf(x0, x1);
      v = fmaxf(v, __shfl_xor(v, 1, 32));
      v = fmaxf(v, __shfl_xor(v, 2, 32));
      v = fmaxf(v, __shfl_xor(v, 4, 32));
      v = fmaxf(v, __shfl_xor(v, 8, 32));
      const float mnew = fmaxf(mr[r], v);
      const float corr = exp2f(mr[r] - mnew);
      const float p0 = exp2f(x0 - mnew);
      const float p1 = exp2f(x1 - mnew);
      float rs = p0 + p1;
      rs += __shfl_xor(rs, 1, 32);
      rs += __shfl_xor(rs, 2, 32);
      rs += __shfl_xor(rs, 4, 32);
      rs += __shfl_xor(rs, 8, 32);
      lr[r] = lr[r] * corr + rs;
      mr[r] = mnew;
      o0[r] *= corr; o1[r] *= corr; o2[r] *= corr; o3[r] *= corr;
      pshm[wave][r + hi * 8][ln]      = (bf16_t)p0;   // C-layout -> (M,N) in LDS
      pshm[wave][r + hi * 8][16 + ln] = (bf16_t)p1;
    }
    // wave-private LDS round trip: same-wave DS ops are in-order; just wait.
    asm volatile("s_wait_dscnt 0x0" ::: "memory");
    __builtin_amdgcn_wave_barrier();
    // ---- re-load P as an A fragment --------------------------------------
    v16bf aP;
    {
      const bf16_t* prow = &pshm[wave][ln][0];
      v8bf c0 = *(const v8bf*)(prow + hi * 16);
      v8bf c1 = *(const v8bf*)(prow + hi * 16 + 8);
#pragma unroll
      for (int i = 0; i < 8; ++i) { aP[i] = c0[i]; aP[i + 8] = c1[i]; }
    }
    // ---- O += P V  (V stored transposed [hd, s]) -------------------------
    {
      v16bf bv0 = *(const v16bf*)(Vh + (long long)(0 * 16 + ln) * S_ + j0 + hi * 16);
      v16bf bv1 = *(const v16bf*)(Vh + (long long)(1 * 16 + ln) * S_ + j0 + hi * 16);
      v16bf bv2 = *(const v16bf*)(Vh + (long long)(2 * 16 + ln) * S_ + j0 + hi * 16);
      v16bf bv3 = *(const v16bf*)(Vh + (long long)(3 * 16 + ln) * S_ + j0 + hi * 16);
      o0 = WMMA_BF16(aP, bv0, o0);
      o1 = WMMA_BF16(aP, bv1, o1);
      o2 = WMMA_BF16(aP, bv2, o2);
      o3 = WMMA_BF16(aP, bv3, o3);
    }
  }
  // ---- normalize + store [b, s, h*64 + hd] as bf16 -------------------------
#pragma unroll
  for (int r = 0; r < 8; ++r) {
    const float inv = lr[r] > 0.f ? 1.f / lr[r] : 0.f;
    const int row = qbase + r + hi * 8;
    bf16_t* dst = AO + ((long long)bb * S_ + row) * E_ + h * HD_;
    dst[0 * 16 + ln] = (bf16_t)(o0[r] * inv);
    dst[1 * 16 + ln] = (bf16_t)(o1[r] * inv);
    dst[2 * 16 + ln] = (bf16_t)(o2[r] * inv);
    dst[3 * 16 + ln] = (bf16_t)(o3[r] * inv);
  }
}

// ------------------------------- projection GEMM ---------------------------
__global__ __launch_bounds__(256) void k_proj_gemm(
    const bf16_t* __restrict__ AO, const bf16_t* __restrict__ WT,
    const float* __restrict__ bias, float* __restrict__ out) {
  const int lane = threadIdx.x & 31, wave = threadIdx.x >> 5;
  const int ln = lane & 15, hi = lane >> 4;
  const int Mbase = blockIdx.y * 128 + wave * 16;
  const int Nbase = blockIdx.x * 128;

  __shared__ bf16_t Bsh[2][TBUF_HALFS_];
  const unsigned int lds_b[2] = {(unsigned int)(size_t)&Bsh[0][0],
                                 (unsigned int)(size_t)&Bsh[1][0]};
  const bf16_t* wtile = WT + (long long)Nbase * E_;

  v8f acc[8];
#pragma unroll
  for (int f = 0; f < 8; ++f) acc[f] = zero_v8f();

  if (wave == 0) tdm_load_tile(lds_b[0], wtile, (unsigned int)E_);

  constexpr int NK = E_ / 32;
  for (int kt = 0; kt < NK; ++kt) {
    if (wave == 0) __builtin_amdgcn_s_wait_tensorcnt(0);
    __syncthreads();
    if (wave == 0 && kt + 1 < NK)
      tdm_load_tile(lds_b[(kt + 1) & 1], wtile + (kt + 1) * 32, (unsigned int)E_);

    v16bf a = load_a_frag(AO + (long long)Mbase * E_ + kt * 32, E_, lane);
    const bf16_t* bsh = &Bsh[kt & 1][0];
#pragma unroll
    for (int f = 0; f < 8; ++f) {
      v16bf bfrag = *(const v16bf*)(bsh + (f * 16 + ln) * TROW_ + hi * 16);
      acc[f] = WMMA_BF16(a, bfrag, acc[f]);
    }
  }
#pragma unroll
  for (int f = 0; f < 8; ++f) {
    const int n = Nbase + f * 16 + ln;
    const float bs = bias[n];
#pragma unroll
    for (int r = 0; r < 8; ++r) {
      const int row = Mbase + r + hi * 8;
      out[(long long)row * E_ + n] = acc[f][r] + bs;  // 16 lanes -> 64B contig
    }
  }
}

// ------------------------------- launcher ----------------------------------
extern "C" void kernel_launch(void* const* d_in, const int* in_sizes, int n_in,
                              void* d_out, int out_size, void* d_ws, size_t ws_size,
                              hipStream_t stream) {
  (void)in_sizes; (void)n_in; (void)out_size; (void)ws_size;
  const float* hidden = (const float*)d_in[0];
  const int*   amask  = (const int*)d_in[1];
  const float* Wqkv   = (const float*)d_in[2];
  const float* bqkv   = (const float*)d_in[3];
  const float* Wproj  = (const float*)d_in[4];
  const float* bproj  = (const float*)d_in[5];
  float* out = (float*)d_out;

  char* ws = (char*)d_ws;
  size_t off = 0;
  auto alloc = [&](size_t bytes) -> char* {
    char* p = ws + off;
    off += (bytes + 255) & ~(size_t)255;
    return p;
  };
  bf16_t* Xb     = (bf16_t*)alloc((size_t)B_ * S_ * E_ * 2);       // 8 MB
  bf16_t* WqkvT  = (bf16_t*)alloc((size_t)3 * E_ * E_ * 2);        // 6 MB
  bf16_t* WprojT = (bf16_t*)alloc((size_t)E_ * E_ * 2);            // 2 MB
  bf16_t* Qp     = (bf16_t*)alloc((size_t)B_ * H_ * S_ * HD_ * 2); // 8 MB
  bf16_t* Kp     = (bf16_t*)alloc((size_t)B_ * H_ * S_ * HD_ * 2); // 8 MB
  bf16_t* Vt     = (bf16_t*)alloc((size_t)B_ * H_ * S_ * HD_ * 2); // 8 MB
  bf16_t* AO     = (bf16_t*)alloc((size_t)B_ * S_ * E_ * 2);       // 8 MB

  {
    const int n4 = B_ * S_ * E_ / 4;
    k_cast_x<<<(n4 + 255) / 256, 256, 0, stream>>>((const float4*)hidden, (v4bf*)Xb, n4);
  }
  k_transpose_cast<<<2048, 256, 0, stream>>>(Wqkv, WqkvT, E_, 3 * E_);
  k_transpose_cast<<<1024, 256, 0, stream>>>(Wproj, WprojT, E_, E_);
  {
    dim3 g(3 * E_ / 128, (B_ * S_) / 128);
    k_qkv_gemm<<<g, 256, 0, stream>>>(Xb, WqkvT, bqkv, Qp, Kp, Vt);
  }
  {
    dim3 g(S_ / 64, B_ * H_);
    k_attn<<<g, 128, 0, stream>>>(Qp, Kp, Vt, amask, AO);
  }
  {
    dim3 g(E_ / 128, (B_ * S_) / 128);
    k_proj_gemm<<<g, 256, 0, stream>>>(AO, WprojT, bproj, out);
  }
}
